// GHGEAT_NoGH_75471165325601
// MI455X (gfx1250) — matrix operand order repaired and checked by reading, not compiled, not run on hardware
//
#include <hip/hip_runtime.h>

#define N_NODES 50000
#define N_EDGES 800000
#define HDIM 128
#define EINCH 64
#define UINCH 3
#define NGRAPH 256
#define MSLOT 128

typedef __attribute__((ext_vector_type(16))) __bf16 bf16x16;
typedef __attribute__((ext_vector_type(8)))  float  f32x8;

union BFragU { bf16x16 v; uint4 q[2]; };

// A-fragment load from row-major LDS tile a[16][kpad] (bf16), per CDNA5 ISA
// 16-bit A 16x32 layout: lane holds row (lane&15); element e maps to
// K = (e&7) + ((e>>3)<<4) + (lane>=16 ? 8 : 0)  -> two contiguous 16B chunks.
__device__ __forceinline__ bf16x16 load_a_frag(const __bf16* base, int kpad, int kt, int lane) {
  int m   = lane & 15;
  int hi8 = (lane & 16) ? 8 : 0;
  const __bf16* p = base + m * kpad + kt * 32 + hi8;
  BFragU f;
  f.q[0] = *reinterpret_cast<const uint4*>(p);        // K = kt*32+hi8 .. +8
  f.q[1] = *reinterpret_cast<const uint4*>(p + 16);   // K = kt*32+16+hi8 .. +8
  return f.v;
}

// B-fragment load from pre-swizzled fragment buffer: 512 bf16 per 16x32 tile,
// lane-major (lane*16 contiguous elements).
__device__ __forceinline__ bf16x16 load_b_frag(const __bf16* frag, int lane) {
  BFragU f;
  const uint4* p = reinterpret_cast<const uint4*>(frag + lane * 16);
  f.q[0] = p[0];
  f.q[1] = p[1];
  return f.v;
}

__device__ __forceinline__ f32x8 wmma_bf16(bf16x16 a, bf16x16 b, f32x8 c) {
  return __builtin_amdgcn_wmma_f32_16x16x32_bf16(false, a, false, b, (short)0, c, false, false);
}

__device__ __forceinline__ void atomicMaxF(float* addr, float val) {
  int* ia  = (int*)addr;
  int  old = __float_as_int(*addr);
  while (__int_as_float(old) < val) {
    int assumed = old;
    old = atomicCAS(ia, assumed, __float_as_int(val));
    if (old == assumed) break;
  }
}

// ---------------------------------------------------------------- utilities
__global__ void __launch_bounds__(256) k_fill(float* p, int n, float v) {
  int i = blockIdx.x * 256 + threadIdx.x;
  if (i < n) p[i] = v;
}

__global__ void __launch_bounds__(256) k_cvt_bf16(const float* __restrict__ src,
                                                  __bf16* __restrict__ dst, int n) {
  int i = blockIdx.x * 256 + threadIdx.x;
  if (i < n) dst[i] = (__bf16)src[i];
}

// Build WMMA B-fragments (bf16), K padded to multiples of 32 with zeros.
// transpose=1: B[k][n] = W[n*cols + k]  (W is (outs=rows, in=cols), e.g. eW1)
// transpose=0: B[k][n] = W[k*cols + n]  (W is (K=rows, N=cols), e.g. Mv)
// B-fragment 16-bit 32x16 layout: lane holds col (lane&15);
// element e -> K = kt*32 + (e&7) + ((e>>3)<<4) + (lane>=16 ? 8 : 0).
__global__ void __launch_bounds__(512) k_prep_frag(const float* __restrict__ W,
    int rows, int cols, int kT, int transpose, __bf16* __restrict__ out) {
  int tileIdx = blockIdx.x;            // nt*kT + kt
  int kt = tileIdx % kT;
  int t  = threadIdx.x;                // 0..511
  int lane = t >> 4, e = t & 15;
  int k = kt * 32 + ((lane & 16) ? 8 : 0) + (e & 7) + ((e >> 3) << 4);
  int n = (tileIdx / kT) * 16 + (lane & 15);
  float v = 0.0f;
  if (transpose) { if (n < rows && k < cols) v = W[(size_t)n * cols + k]; }
  else           { if (k < rows && n < cols) v = W[(size_t)k * cols + n]; }
  out[(size_t)tileIdx * 512 + t] = (__bf16)v;
}

// ---------------------------------------------------------------- EdgeModel
// e_cat = [x[row](128), x[col](128), edge_attr(64), u[batch[row]](3)] -> 323 (pad 352)
// e_new = (relu(e_cat @ eW1.T + eb1)) @ eW2.T + eb2 ; scatter-add into agg[col]
__global__ void __launch_bounds__(256) k_edge(
    const __bf16* __restrict__ xb, const float* __restrict__ edge_attr,
    const float* __restrict__ u,   const int* __restrict__ edge_index,
    const int* __restrict__ batch,
    const __bf16* __restrict__ fe1, const float* __restrict__ eb1,
    const __bf16* __restrict__ fe2, const float* __restrict__ eb2,
    float* __restrict__ eout, float* __restrict__ agg, float* __restrict__ deg) {
  __shared__ __align__(16) __bf16 sA[16 * 352];
  __shared__ __align__(16) __bf16 sH[16 * 128];
  __shared__ int sRow[16], sCol[16];
  const int tid = threadIdx.x;
  const int e0  = blockIdx.x * 16;
  if (tid < 16) {
    sRow[tid] = edge_index[e0 + tid];
    sCol[tid] = edge_index[N_EDGES + e0 + tid];
  }
  __syncthreads();
  for (int idx = tid; idx < 16 * 352; idx += 256) {
    int m = idx / 352, c = idx - m * 352;
    __bf16 v = (__bf16)0.0f;
    if (c < 128)      v = xb[(size_t)sRow[m] * HDIM + c];
    else if (c < 256) v = xb[(size_t)sCol[m] * HDIM + (c - 128)];
    else if (c < 320) v = (__bf16)edge_attr[(size_t)(e0 + m) * EINCH + (c - 256)];
    else if (c < 323) v = (__bf16)u[batch[sRow[m]] * UINCH + (c - 320)];
    sA[idx] = v;
  }
  __syncthreads();
  const int wave = tid >> 5, lane = tid & 31;
  const int n   = (wave << 4) + (lane & 15);
  const int hi8 = (lane & 16) ? 8 : 0;
  f32x8 acc = {};
  for (int kt = 0; kt < 11; ++kt)
    acc = wmma_bf16(load_a_frag(sA, 352, kt, lane),
                    load_b_frag(fe1 + (size_t)(wave * 11 + kt) * 512, lane), acc);
  {
    float b = eb1[n];
#pragma unroll
    for (int r = 0; r < 8; ++r) {
      float h = acc[r] + b;
      sH[(hi8 + r) * 128 + n] = (__bf16)(h > 0.0f ? h : 0.0f);
    }
  }
  __syncthreads();
  f32x8 acc2 = {};
  for (int kt = 0; kt < 4; ++kt)
    acc2 = wmma_bf16(load_a_frag(sH, 128, kt, lane),
                     load_b_frag(fe2 + (size_t)(wave * 4 + kt) * 512, lane), acc2);
  {
    float b = eb2[n];
#pragma unroll
    for (int r = 0; r < 8; ++r) {
      int m = hi8 + r;
      float v = acc2[r] + b;
      eout[(size_t)(e0 + m) * HDIM + n] = v;
      atomicAdd(&agg[(size_t)sCol[m] * HDIM + n], v);
    }
  }
  if (tid < 16) atomicAdd(&deg[sCol[tid]], 1.0f);
}

// ---------------------------------------------------------------- attn logits
// combined = [x(128), agg(128), u[batch](3)] -> 259 (pad 288); logits = combined @ Mk.T
__global__ void __launch_bounds__(256) k_logits(
    const __bf16* __restrict__ xb, const float* __restrict__ agg,
    const float* __restrict__ u,   const int* __restrict__ batch,
    const __bf16* __restrict__ fmk, float* __restrict__ logits,
    float* __restrict__ colmax) {
  __shared__ __align__(16) __bf16 sA[16 * 288];
  const int tid = threadIdx.x;
  const int i0  = blockIdx.x * 16;
  for (int idx = tid; idx < 16 * 288; idx += 256) {
    int m = idx / 288, c = idx - m * 288;
    int i = i0 + m;
    __bf16 v = (__bf16)0.0f;
    if (c < 128)      v = xb[(size_t)i * HDIM + c];
    else if (c < 256) v = (__bf16)agg[(size_t)i * HDIM + (c - 128)];
    else if (c < 259) v = (__bf16)u[batch[i] * UINCH + (c - 256)];
    sA[idx] = v;
  }
  __syncthreads();
  const int wave = tid >> 5, lane = tid & 31;
  const int n   = (wave << 4) + (lane & 15);
  const int hi8 = (lane & 16) ? 8 : 0;
  f32x8 acc = {};
  for (int kt = 0; kt < 9; ++kt)
    acc = wmma_bf16(load_a_frag(sA, 288, kt, lane),
                    load_b_frag(fmk + (size_t)(wave * 9 + kt) * 512, lane), acc);
  float lm = -3.0e38f;
#pragma unroll
  for (int r = 0; r < 8; ++r) {
    logits[(size_t)(i0 + hi8 + r) * MSLOT + n] = acc[r];
    lm = fmaxf(lm, acc[r]);
  }
  lm = fmaxf(lm, __shfl_xor(lm, 16, 32));  // combine lanes n and n+16 (same column)
  if (lane < 16) atomicMaxF(&colmax[n], lm);
}

// exp(l - colmax[c]) in place; accumulate column sums via LDS partials
__global__ void __launch_bounds__(256) k_softexp(float* __restrict__ logits,
    const float* __restrict__ colmax, float* __restrict__ colsum) {
  __shared__ float part[MSLOT];
  const int tid = threadIdx.x;
  if (tid < MSLOT) part[tid] = 0.0f;
  __syncthreads();
  const size_t base = (size_t)blockIdx.x * 16 * MSLOT;
  for (int idx = tid; idx < 16 * MSLOT; idx += 256) {
    int c = idx & (MSLOT - 1);
    float v = __expf(logits[base + idx] - colmax[c]);
    logits[base + idx] = v;
    atomicAdd(&part[c], v);
  }
  __syncthreads();
  if (tid < MSLOT) atomicAdd(&colsum[tid], part[tid]);
}

// attn = (exp/colsum); L1-normalize each row; store bf16 (one wave per row)
__global__ void __launch_bounds__(256) k_attnnorm(const float* __restrict__ logits,
    const float* __restrict__ colsum, __bf16* __restrict__ attnb) {
  const int tid = threadIdx.x;
  const int wave = tid >> 5, lane = tid & 31;
  const size_t i = (size_t)blockIdx.x * 8 + wave;
  const float* row = logits + i * MSLOT;
  float w0 = row[lane]      / colsum[lane];
  float w1 = row[lane + 32] / colsum[lane + 32];
  float w2 = row[lane + 64] / colsum[lane + 64];
  float w3 = row[lane + 96] / colsum[lane + 96];
  float s = w0 + w1 + w2 + w3;
  for (int off = 16; off > 0; off >>= 1) s += __shfl_xor(s, off, 32);
  float inv = 1.0f / s;
  __bf16* orow = attnb + i * MSLOT;
  orow[lane]      = (__bf16)(w0 * inv);
  orow[lane + 32] = (__bf16)(w1 * inv);
  orow[lane + 64] = (__bf16)(w2 * inv);
  orow[lane + 96] = (__bf16)(w3 * inv);
}

// edge_agg sums derived per node: edge_sum[b] += agg[i]; edge_cnt[b] += deg[i]
__global__ void __launch_bounds__(256) k_edge_agg(
    const float* __restrict__ agg, const float* __restrict__ deg,
    const int* __restrict__ batch, float* __restrict__ edge_sum,
    float* __restrict__ edge_cnt) {
  size_t idx = (size_t)blockIdx.x * 256 + threadIdx.x;
  if (idx >= (size_t)N_NODES * HDIM) return;
  int i = (int)(idx >> 7), c = (int)(idx & 127);
  int b = batch[i];
  atomicAdd(&edge_sum[b * HDIM + c], agg[idx]);
  if (c == 0) atomicAdd(&edge_cnt[b], deg[i]);
}

// ---------------------------------------------------------------- NodeModel
// attn_out = attn @ Mv (259 cols, pad 288); x_new = MLP(attn_out); node mean accum
__global__ void __launch_bounds__(256) k_node(
    const __bf16* __restrict__ attnb, const int* __restrict__ batch,
    const __bf16* __restrict__ fmv,
    const __bf16* __restrict__ fn1, const float* __restrict__ nb1,
    const __bf16* __restrict__ fn2, const float* __restrict__ nb2,
    float* __restrict__ xout, float* __restrict__ node_sum,
    float* __restrict__ node_cnt) {
  __shared__ __align__(16) __bf16 sAO[16 * 288];
  __shared__ __align__(16) __bf16 sAH[16 * 128];  // attn tile, later hidden tile
  const int tid = threadIdx.x;
  const int i0  = blockIdx.x * 16;
  for (int idx = tid; idx < 16 * 128; idx += 256)
    sAH[idx] = attnb[(size_t)i0 * MSLOT + idx];
  for (int idx = tid; idx < 16 * 288; idx += 256) sAO[idx] = (__bf16)0.0f;
  __syncthreads();
  const int wave = tid >> 5, lane = tid & 31;
  const int hi8  = (lane & 16) ? 8 : 0;
  // phase 1: attn_out tiles (17 col-tiles over 8 waves)
  for (int nt = wave; nt < 17; nt += 8) {
    f32x8 acc = {};
    for (int kt = 0; kt < 4; ++kt)
      acc = wmma_bf16(load_a_frag(sAH, 128, kt, lane),
                      load_b_frag(fmv + (size_t)(nt * 4 + kt) * 512, lane), acc);
    int nn = (nt << 4) + (lane & 15);
    if (nn < 259) {
#pragma unroll
      for (int r = 0; r < 8; ++r) sAO[(hi8 + r) * 288 + nn] = (__bf16)acc[r];
    }
  }
  __syncthreads();
  // phase 2: hidden = relu(attn_out @ nW1.T + nb1) -> sAH (safe after barrier)
  const int n = (wave << 4) + (lane & 15);
  f32x8 acc = {};
  for (int kt = 0; kt < 9; ++kt)
    acc = wmma_bf16(load_a_frag(sAO, 288, kt, lane),
                    load_b_frag(fn1 + (size_t)(wave * 9 + kt) * 512, lane), acc);
  {
    float b = nb1[n];
#pragma unroll
    for (int r = 0; r < 8; ++r) {
      float h = acc[r] + b;
      sAH[(hi8 + r) * 128 + n] = (__bf16)(h > 0.0f ? h : 0.0f);
    }
  }
  __syncthreads();
  // phase 3: x_new = hidden @ nW2.T + nb2
  f32x8 acc2 = {};
  for (int kt = 0; kt < 4; ++kt)
    acc2 = wmma_bf16(load_a_frag(sAH, 128, kt, lane),
                     load_b_frag(fn2 + (size_t)(wave * 4 + kt) * 512, lane), acc2);
  {
    float b = nb2[n];
#pragma unroll
    for (int r = 0; r < 8; ++r) {
      int m = hi8 + r;
      float v = acc2[r] + b;
      xout[(size_t)(i0 + m) * HDIM + n] = v;
      atomicAdd(&node_sum[batch[i0 + m] * HDIM + n], v);
    }
  }
  if (tid < 16) atomicAdd(&node_cnt[batch[i0 + tid]], 1.0f);
}

// ---------------------------------------------------------------- GlobalModel
__global__ void __launch_bounds__(256) k_global(
    const float* __restrict__ u,
    const float* __restrict__ node_sum, const float* __restrict__ node_cnt,
    const float* __restrict__ edge_sum, const float* __restrict__ edge_cnt,
    const __bf16* __restrict__ fg1, const float* __restrict__ gb1,
    const __bf16* __restrict__ fg2, const float* __restrict__ gb2,
    float* __restrict__ uout) {
  __shared__ __align__(16) __bf16 sA[16 * 288];
  __shared__ __align__(16) __bf16 sH[16 * 128];
  const int tid = threadIdx.x;
  const int g0  = blockIdx.x * 16;
  for (int idx = tid; idx < 16 * 288; idx += 256) {
    int m = idx / 288, c = idx - m * 288;
    int g = g0 + m;
    float v = 0.0f;
    if (c < 3)        v = u[g * UINCH + c];
    else if (c < 131) v = node_sum[g * HDIM + (c - 3)]   / fmaxf(node_cnt[g], 1.0f);
    else if (c < 259) v = edge_sum[g * HDIM + (c - 131)] / fmaxf(edge_cnt[g], 1.0f);
    sA[idx] = (__bf16)v;
  }
  __syncthreads();
  const int wave = tid >> 5, lane = tid & 31;
  const int n   = (wave << 4) + (lane & 15);
  const int hi8 = (lane & 16) ? 8 : 0;
  f32x8 acc = {};
  for (int kt = 0; kt < 9; ++kt)
    acc = wmma_bf16(load_a_frag(sA, 288, kt, lane),
                    load_b_frag(fg1 + (size_t)(wave * 9 + kt) * 512, lane), acc);
  {
    float b = gb1[n];
#pragma unroll
    for (int r = 0; r < 8; ++r) {
      float h = acc[r] + b;
      sH[(hi8 + r) * 128 + n] = (__bf16)(h > 0.0f ? h : 0.0f);
    }
  }
  __syncthreads();
  f32x8 acc2 = {};
  for (int kt = 0; kt < 4; ++kt)
    acc2 = wmma_bf16(load_a_frag(sH, 128, kt, lane),
                     load_b_frag(fg2 + (size_t)(wave * 4 + kt) * 512, lane), acc2);
  {
    float b = gb2[n];
#pragma unroll
    for (int r = 0; r < 8; ++r)
      uout[(size_t)(g0 + hi8 + r) * HDIM + n] = acc2[r] + b;
  }
}

// ---------------------------------------------------------------- host launch
extern "C" void kernel_launch(void* const* d_in, const int* in_sizes, int n_in,
                              void* d_out, int out_size, void* d_ws, size_t ws_size,
                              hipStream_t stream) {
  const float* x          = (const float*)d_in[0];
  const float* edge_attr  = (const float*)d_in[1];
  const float* u          = (const float*)d_in[2];
  const int*   edge_index = (const int*)d_in[3];
  const int*   batch      = (const int*)d_in[4];
  const float* eW1 = (const float*)d_in[5];
  const float* eb1 = (const float*)d_in[6];
  const float* eW2 = (const float*)d_in[7];
  const float* eb2 = (const float*)d_in[8];
  const float* Mk  = (const float*)d_in[9];
  const float* Mv  = (const float*)d_in[10];
  const float* nW1 = (const float*)d_in[11];
  const float* nb1 = (const float*)d_in[12];
  const float* nW2 = (const float*)d_in[13];
  const float* nb2 = (const float*)d_in[14];
  const float* gW1 = (const float*)d_in[15];
  const float* gb1 = (const float*)d_in[16];
  const float* gW2 = (const float*)d_in[17];
  const float* gb2 = (const float*)d_in[18];

  float* xout = (float*)d_out;
  float* eout = xout + (size_t)N_NODES * HDIM;
  float* uout = eout + (size_t)N_EDGES * HDIM;

  char* ws = (char*)d_ws;
  size_t off = 0;
  auto alloc = [&](size_t bytes) -> char* {
    char* p = ws + off;
    off = (off + bytes + 255) & ~(size_t)255;
    return p;
  };
  __bf16* xb       = (__bf16*)alloc((size_t)N_NODES * HDIM * 2);
  float*  agg      = (float*) alloc((size_t)N_NODES * HDIM * 4);
  float*  logits   = (float*) alloc((size_t)N_NODES * MSLOT * 4);
  __bf16* attnb    = (__bf16*)alloc((size_t)N_NODES * MSLOT * 2);
  float*  deg      = (float*) alloc((size_t)N_NODES * 4);
  float*  colmax   = (float*) alloc(MSLOT * 4);
  float*  colsum   = (float*) alloc(MSLOT * 4);
  float*  node_sum = (float*) alloc(NGRAPH * HDIM * 4);
  float*  node_cnt = (float*) alloc(NGRAPH * 4);
  float*  edge_sum = (float*) alloc(NGRAPH * HDIM * 4);
  float*  edge_cnt = (float*) alloc(NGRAPH * 4);
  __bf16* fe1 = (__bf16*)alloc((size_t)8 * 11 * 512 * 2);
  __bf16* fe2 = (__bf16*)alloc((size_t)8 * 4  * 512 * 2);
  __bf16* fmk = (__bf16*)alloc((size_t)8 * 9  * 512 * 2);
  __bf16* fmv = (__bf16*)alloc((size_t)17 * 4 * 512 * 2);
  __bf16* fn1 = (__bf16*)alloc((size_t)8 * 9  * 512 * 2);
  __bf16* fn2 = (__bf16*)alloc((size_t)8 * 4  * 512 * 2);
  __bf16* fg1 = (__bf16*)alloc((size_t)8 * 9  * 512 * 2);
  __bf16* fg2 = (__bf16*)alloc((size_t)8 * 4  * 512 * 2);

  // deterministic init of accumulators
  k_fill<<<(N_NODES * HDIM + 255) / 256, 256, 0, stream>>>(agg, N_NODES * HDIM, 0.f);
  k_fill<<<(N_NODES + 255) / 256, 256, 0, stream>>>(deg, N_NODES, 0.f);
  k_fill<<<1, 256, 0, stream>>>(colmax, MSLOT, -3.0e38f);
  k_fill<<<1, 256, 0, stream>>>(colsum, MSLOT, 0.f);
  k_fill<<<(NGRAPH * HDIM + 255) / 256, 256, 0, stream>>>(node_sum, NGRAPH * HDIM, 0.f);
  k_fill<<<1, 256, 0, stream>>>(node_cnt, NGRAPH, 0.f);
  k_fill<<<(NGRAPH * HDIM + 255) / 256, 256, 0, stream>>>(edge_sum, NGRAPH * HDIM, 0.f);
  k_fill<<<1, 256, 0, stream>>>(edge_cnt, NGRAPH, 0.f);

  // precision prep: x -> bf16; weights -> swizzled bf16 WMMA B-fragments
  k_cvt_bf16<<<(N_NODES * HDIM + 255) / 256, 256, 0, stream>>>(x, xb, N_NODES * HDIM);
  k_prep_frag<<<8 * 11, 512, 0, stream>>>(eW1, HDIM, 2 * HDIM + EINCH + UINCH, 11, 1, fe1);
  k_prep_frag<<<8 * 4,  512, 0, stream>>>(eW2, HDIM, HDIM, 4, 1, fe2);
  k_prep_frag<<<8 * 9,  512, 0, stream>>>(Mk,  MSLOT, 2 * HDIM + UINCH, 9, 1, fmk);
  k_prep_frag<<<17 * 4, 512, 0, stream>>>(Mv,  MSLOT, 2 * HDIM + UINCH, 4, 0, fmv);
  k_prep_frag<<<8 * 9,  512, 0, stream>>>(nW1, HDIM, 2 * HDIM + UINCH, 9, 1, fn1);
  k_prep_frag<<<8 * 4,  512, 0, stream>>>(nW2, HDIM, HDIM, 4, 1, fn2);
  k_prep_frag<<<8 * 9,  512, 0, stream>>>(gW1, HDIM, 2 * HDIM + UINCH, 9, 1, fg1);
  k_prep_frag<<<8 * 4,  512, 0, stream>>>(gW2, HDIM, HDIM, 4, 1, fg2);

  // pipeline
  k_edge<<<N_EDGES / 16, 256, 0, stream>>>(xb, edge_attr, u, edge_index, batch,
                                           fe1, eb1, fe2, eb2, eout, agg, deg);
  k_logits<<<N_NODES / 16, 256, 0, stream>>>(xb, agg, u, batch, fmk, logits, colmax);
  k_softexp<<<N_NODES / 16, 256, 0, stream>>>(logits, colmax, colsum);
  k_attnnorm<<<N_NODES / 8, 256, 0, stream>>>(logits, colsum, attnb);
  k_edge_agg<<<(N_NODES * HDIM) / 256, 256, 0, stream>>>(agg, deg, batch, edge_sum, edge_cnt);
  k_node<<<N_NODES / 16, 256, 0, stream>>>(attnb, batch, fmv, fn1, nb1, fn2, nb2,
                                           xout, node_sum, node_cnt);
  k_global<<<NGRAPH / 16, 256, 0, stream>>>(u, node_sum, node_cnt, edge_sum, edge_cnt,
                                            fg1, gb1, fg2, gb2, uout);
}
